// RGT_18013092839760
// MI455X (gfx1250) — matrix operand order repaired and checked by reading, not compiled, not run on hardware
//
#include <hip/hip_runtime.h>
#include <hip/hip_bf16.h>
#include <math.h>

// Problem constants (match reference)
#define NN   30000
#define IND  768
#define CD   128
#define NH   4
#define DHD  128
#define HDD  512   // NH*DHD
#define RR   3
#define EE   80000
#define NEGS 0.01f

typedef __attribute__((ext_vector_type(16))) __bf16          v16bf;
typedef __attribute__((ext_vector_type(8)))  float           v8f;
typedef __attribute__((ext_vector_type(8)))  unsigned short  us8;   // 16 bytes

union Frag { unsigned short u[16]; us8 h[2]; v16bf v; };

__device__ __forceinline__ unsigned short f2bf(float f) {
  unsigned u = __float_as_uint(f);
  unsigned r = 0x7FFFu + ((u >> 16) & 1u);
  return (unsigned short)((u + r) >> 16);
}

// ---------------- utility kernels ----------------
__global__ void k_fill(float* __restrict__ p, float v, size_t n) {
  size_t i = (size_t)blockIdx.x * blockDim.x + threadIdx.x;
  if (i < n) p[i] = v;
}

__global__ void k_cvt(const float* __restrict__ in, unsigned short* __restrict__ out, size_t n) {
  size_t i = (size_t)blockIdx.x * blockDim.x + threadIdx.x;
  if (i < n) out[i] = f2bf(in[i]);
}

// Pack f32 B[K,Nc] (row-major) into bf16 WMMA-B fragment order:
// Bp[((k/32)*Nc + n)*32 + (k%32)] = bf16(B[k*Nc + n])
// -> per column, each 32-k block is 64 contiguous bytes; a lane reads two 16-byte chunks of it.
__global__ void k_pack_b(const float* __restrict__ B, unsigned short* __restrict__ Bp,
                         int K, int Nc) {
  size_t id = (size_t)blockIdx.x * blockDim.x + threadIdx.x;
  if (id >= (size_t)K * Nc) return;
  const int t  = (int)(id & 31);
  const int n  = (int)((id >> 5) % Nc);
  const int kt = (int)(id / ((size_t)32 * Nc));
  const int k  = kt * 32 + t;
  Bp[id] = f2bf(B[(size_t)k * Nc + n]);
}

// ---------------- WMMA GEMM: C[M,Nc] = act(A[M,K]*Bp + bias) ----------------
// A bf16 row-major (16B-aligned rows); Bp packed by k_pack_b; C f32 row-major.
// Each wave computes a 16x64 strip (4 independent accumulators -> no WMMA RAW
// hazard NOPs, A fragment reused 4x). 4 waves/block (64 rows). M % 16 == 0,
// Nc % 64 == 0 required.
template <int K, int ACT>
__global__ void k_wmma_gemm(const unsigned short* __restrict__ A,
                            const unsigned short* __restrict__ Bp,
                            const float* __restrict__ bias,
                            float* __restrict__ Cm, int M, int Nc) {
  const int lane = threadIdx.x & 31;
  const int wave = threadIdx.x >> 5;
  const int hl   = lane >> 4;       // half-wave select
  const int q    = lane & 15;
  const int mBase = blockIdx.x * 64 + wave * 16;
  const int nBase = blockIdx.y * 64;
  if (mBase >= M) return;           // wave-uniform: EXEC stays all-ones for WMMA

  const unsigned short* Arow = A + (size_t)(mBase + q) * K;
  v8f acc[4] = {{}, {}, {}, {}};
#pragma unroll
  for (int kb = 0; kb < K; kb += 32) {
    Frag a;
    // A 16x32 layout: lanes 0-15 hold K{0..7,16..23}; lanes 16-31 K{8..15,24..31}
    a.h[0] = *(const us8*)(Arow + kb + 8 * hl);
    a.h[1] = *(const us8*)(Arow + kb + 16 + 8 * hl);
    // B 32x16 layout: lanes 0-15 hold K{0..7,16..23}; lanes 16-31 K{8..15,24..31} (packed)
    const unsigned short* bp = Bp + ((size_t)(kb >> 5) * Nc + nBase + q) * 32 + 8 * hl;
    Frag b0, b1, b2, b3;
    b0.h[0] = *(const us8*)(bp);          b0.h[1] = *(const us8*)(bp + 16);
    b1.h[0] = *(const us8*)(bp + 512);    b1.h[1] = *(const us8*)(bp + 528);
    b2.h[0] = *(const us8*)(bp + 1024);   b2.h[1] = *(const us8*)(bp + 1040);
    b3.h[0] = *(const us8*)(bp + 1536);   b3.h[1] = *(const us8*)(bp + 1552);
    acc[0] = __builtin_amdgcn_wmma_f32_16x16x32_bf16(false, a.v, false, b0.v,
                                                     (short)0, acc[0], false, false);
    acc[1] = __builtin_amdgcn_wmma_f32_16x16x32_bf16(false, a.v, false, b1.v,
                                                     (short)0, acc[1], false, false);
    acc[2] = __builtin_amdgcn_wmma_f32_16x16x32_bf16(false, a.v, false, b2.v,
                                                     (short)0, acc[2], false, false);
    acc[3] = __builtin_amdgcn_wmma_f32_16x16x32_bf16(false, a.v, false, b3.v,
                                                     (short)0, acc[3], false, false);
  }
#pragma unroll
  for (int t = 0; t < 4; ++t) {
    const int col = nBase + 16 * t + q;
    const float bv = bias[col];
#pragma unroll
    for (int j = 0; j < 8; ++j) {
      float val = acc[t][j] + bv;          // C/D: VGPR j -> M = j + 8*half
      if (ACT == 1) val = (val >= 0.0f) ? val : NEGS * val;
      Cm[(size_t)(mBase + j + 8 * hl) * Nc + col] = val;
    }
  }
}

// ---------------- edge (scatter-softmax) kernels ----------------
__device__ __forceinline__ void atomicMaxF(float* a, float v) {
  if (v >= 0.0f) atomicMax((int*)a, __float_as_int(v));
  else atomicMin((unsigned int*)a, (unsigned int)__float_as_int(v));
}

__global__ void k_edge_logits(const int* __restrict__ src, const int* __restrict__ dst,
                              const float* __restrict__ qm, const float* __restrict__ km,
                              float* __restrict__ logits, float* __restrict__ segmax) {
  size_t id = (size_t)blockIdx.x * blockDim.x + threadIdx.x;
  if (id >= (size_t)EE * NH) return;
  const int e = (int)(id >> 2), h = (int)(id & 3);
  const int s = src[e], d = dst[e];
  const float4* qp = (const float4*)(qm + (size_t)d * HDD + h * DHD);
  const float4* kp = (const float4*)(km + (size_t)s * HDD + h * DHD);
  float acc = 0.0f;
#pragma unroll 8
  for (int i = 0; i < DHD / 4; ++i) {
    const float4 a = qp[i], b = kp[i];
    acc += a.x * b.x + a.y * b.y + a.z * b.z + a.w * b.w;
  }
  acc *= 0.0883883476483184f; // 1/sqrt(128)
  logits[id] = acc;
  atomicMaxF(segmax + (size_t)d * NH + h, acc);
}

__global__ void k_edge_exp(const int* __restrict__ dst, const float* __restrict__ logits,
                           const float* __restrict__ segmax, float* __restrict__ expo,
                           float* __restrict__ segsum) {
  size_t id = (size_t)blockIdx.x * blockDim.x + threadIdx.x;
  if (id >= (size_t)EE * NH) return;
  const int e = (int)(id >> 2), h = (int)(id & 3);
  const int d = dst[e];
  const float a = __expf(logits[id] - segmax[(size_t)d * NH + h]);
  expo[id] = a;
  atomicAdd(segsum + (size_t)d * NH + h, a);
}

__global__ void k_edge_scatter(const int* __restrict__ src, const int* __restrict__ dst,
                               const float* __restrict__ expo, const float* __restrict__ segsum,
                               const float* __restrict__ vm, float* __restrict__ agg) {
  size_t id = (size_t)blockIdx.x * blockDim.x + threadIdx.x;
  if (id >= (size_t)EE * HDD) return;
  const int e = (int)(id >> 9), c = (int)(id & 511), h = c >> 7;
  const int s = src[e], d = dst[e];
  const float w = expo[(size_t)e * NH + h] / segsum[(size_t)d * NH + h];
  atomicAdd(agg + (size_t)d * HDD + c, w * vm[(size_t)s * HDD + c]);
}

// U[n,c] += mean over heads of agg
__global__ void k_head_mean(const float* __restrict__ agg, float* __restrict__ U) {
  size_t id = (size_t)blockIdx.x * blockDim.x + threadIdx.x;
  if (id >= (size_t)NN * CD) return;
  const size_t n = id >> 7; const int c = (int)(id & 127);
  const float* a = agg + n * HDD;
  U[id] += 0.25f * (a[c] + a[CD + c] + a[2 * CD + c] + a[3 * CD + c]);
}

// cat = [bf16(U) | h_bf16]  (concat order matches reference: [U, x])
__global__ void k_pack_cat(const float* __restrict__ U, const unsigned short* __restrict__ hb,
                           unsigned short* __restrict__ cat) {
  size_t id = (size_t)blockIdx.x * blockDim.x + threadIdx.x;
  if (id >= (size_t)NN * CD) return;
  const size_t n = id >> 7; const int c = (int)(id & 127);
  cat[n * (CD + DHD) + c]      = f2bf(U[id]);
  cat[n * (CD + DHD) + CD + c] = hb[id];
}

// Hs[n,r,:] = tanh(U)*sigmoid(zlin) + h*(1-sigmoid(zlin))
__global__ void k_gate(const float* __restrict__ zlin, const float* __restrict__ U,
                       const float* __restrict__ h, float* __restrict__ Hs, int r) {
  size_t id = (size_t)blockIdx.x * blockDim.x + threadIdx.x;
  if (id >= (size_t)NN * CD) return;
  const size_t n = id >> 7; const int c = (int)(id & 127);
  const float z = 1.0f / (1.0f + __expf(-zlin[id]));
  Hs[(n * RR + r) * CD + c] = tanhf(U[id]) * z + h[id] * (1.0f - z);
}

// scoresP[h,r] += sum over rows of tanh(T[row,:]) . sw2[h]   (block-local LDS reduce first)
__global__ void k_sem_score(const float* __restrict__ T, const float* __restrict__ sw2h,
                            float* __restrict__ scoresP, int h) {
  __shared__ float part[RR];
  if (threadIdx.x < RR) part[threadIdx.x] = 0.0f;
  __syncthreads();
  const int row = blockIdx.x * blockDim.x + threadIdx.x;
  if (row < NN * RR) {
    const float4* t = (const float4*)(T + (size_t)row * CD);
    const float4* w = (const float4*)sw2h;
    float acc = 0.0f;
#pragma unroll 4
    for (int c = 0; c < CD / 4; ++c) {
      const float4 a = t[c], b = w[c];
      acc += tanhf(a.x) * b.x + tanhf(a.y) * b.y + tanhf(a.z) * b.z + tanhf(a.w) * b.w;
    }
    atomicAdd(&part[row % RR], acc);
  }
  __syncthreads();
  if (threadIdx.x < RR) atomicAdd(scoresP + h * RR + threadIdx.x, part[threadIdx.x]);
}

// softmax over relations per head, averaged -> wr[r] = (1/H) sum_h beta[h,r]
__global__ void k_beta(const float* __restrict__ scoresP, float* __restrict__ wr) {
  if (threadIdx.x != 0 || blockIdx.x != 0) return;
  float w[RR] = {0.0f, 0.0f, 0.0f};
  for (int h = 0; h < NH; ++h) {
    float sc[RR], mx = -3.0e38f;
    for (int r = 0; r < RR; ++r) { sc[r] = scoresP[h * RR + r] / (float)NN; mx = fmaxf(mx, sc[r]); }
    float s = 0.0f;
    for (int r = 0; r < RR; ++r) { sc[r] = __expf(sc[r] - mx); s += sc[r]; }
    for (int r = 0; r < RR; ++r) w[r] += sc[r] / s;
  }
  for (int r = 0; r < RR; ++r) wr[r] = w[r] / (float)NH;
}

__global__ void k_sem_combine(const float* __restrict__ Hs, const float* __restrict__ wr,
                              float* __restrict__ h) {
  size_t id = (size_t)blockIdx.x * blockDim.x + threadIdx.x;
  if (id >= (size_t)NN * CD) return;
  const size_t n = id >> 7; const int c = (int)(id & 127);
  const float* hm = Hs + n * RR * CD + c;
  h[id] = wr[0] * hm[0] + wr[1] * hm[CD] + wr[2] * hm[2 * CD];
}

__global__ void k_clf(const float* __restrict__ h2, const float* __restrict__ W,
                      const float* __restrict__ b, float* __restrict__ out) {
  const int n = blockIdx.x * blockDim.x + threadIdx.x;
  if (n >= NN) return;
  float a0 = b[0], a1 = b[1];
  const float4* hp = (const float4*)(h2 + (size_t)n * CD);
#pragma unroll 4
  for (int c = 0; c < CD / 4; ++c) {
    const float4 x = hp[c];
    a0 += x.x * W[8 * c] + x.y * W[8 * c + 2] + x.z * W[8 * c + 4] + x.w * W[8 * c + 6];
    a1 += x.x * W[8 * c + 1] + x.y * W[8 * c + 3] + x.z * W[8 * c + 5] + x.w * W[8 * c + 7];
  }
  out[2 * n] = a0; out[2 * n + 1] = a1;
}

// ---------------- host orchestration ----------------
static inline unsigned cdivu(size_t a, size_t b) { return (unsigned)((a + b - 1) / b); }

extern "C" void kernel_launch(void* const* d_in, const int* in_sizes, int n_in,
                              void* d_out, int out_size, void* d_ws, size_t ws_size,
                              hipStream_t stream) {
  (void)in_sizes; (void)n_in; (void)out_size; (void)ws_size;
  const float* x      = (const float*)d_in[0];
  const int*   edges  = (const int*)d_in[1];
  const float* lin1_W = (const float*)d_in[2];
  const float* lin1_b = (const float*)d_in[3];
  const float* lin2_W = (const float*)d_in[30];
  const float* lin2_b = (const float*)d_in[31];
  const float* clf_W  = (const float*)d_in[32];
  const float* clf_b  = (const float*)d_in[33];

  // carve workspace (all allocations 64B-aligned for b128 loads)
  char* p = (char*)d_ws;
  auto af = [&](size_t n) { float* r = (float*)p; p += ((n * 4 + 63) & ~(size_t)63); return r; };
  auto ab = [&](size_t n) { unsigned short* r = (unsigned short*)p; p += ((n * 2 + 63) & ~(size_t)63); return r; };

  float* h      = af((size_t)NN * CD);
  float* U      = af((size_t)NN * CD);
  float* qm     = af((size_t)NN * HDD);
  float* km     = af((size_t)NN * HDD);
  float* vm     = af((size_t)NN * HDD);
  float* agg    = af((size_t)NN * HDD);
  float* Hs     = af((size_t)NN * RR * CD);
  float* T      = af((size_t)NN * RR * CD);   // also reused as zlin scratch
  float* logits = af((size_t)EE * NH);
  float* expo   = af((size_t)EE * NH);
  float* segmax = af((size_t)NN * NH);
  float* segsum = af((size_t)NN * NH);
  float* scoresP = af(NH * RR);
  float* wr      = af(4);

  unsigned short* xb     = ab((size_t)NN * IND);
  unsigned short* hb     = ab((size_t)NN * CD);
  unsigned short* catb   = ab((size_t)NN * (CD + DHD));
  unsigned short* Hsb    = ab((size_t)NN * RR * CD);
  unsigned short* lin1Wb = ab((size_t)IND * CD);
  unsigned short* lin2Wb = ab((size_t)CD * CD);
  unsigned short *Wqb[2], *Wkb[2], *Wvb[2], *Wsb[2], *gWb[2], *sW1b[2];
  for (int l = 0; l < 2; ++l) {
    Wqb[l]  = ab((size_t)RR * CD * HDD);
    Wkb[l]  = ab((size_t)RR * CD * HDD);
    Wvb[l]  = ab((size_t)RR * CD * HDD);
    Wsb[l]  = ab((size_t)RR * CD * DHD);
    gWb[l]  = ab((size_t)(CD + DHD) * CD);
    sW1b[l] = ab((size_t)NH * CD * CD);
  }

  auto conv = [&](const float* in, unsigned short* out, size_t n) {
    k_cvt<<<cdivu(n, 256), 256, 0, stream>>>(in, out, n);
  };
  auto fill = [&](float* ptr, float v, size_t n) {
    k_fill<<<cdivu(n, 256), 256, 0, stream>>>(ptr, v, n);
  };
  auto packB = [&](const float* B, unsigned short* Bp, int K, int Nc) {
    k_pack_b<<<cdivu((size_t)K * Nc, 256), 256, 0, stream>>>(B, Bp, K, Nc);
  };
  auto gemm = [&](const unsigned short* A, const unsigned short* Bp, const float* bias,
                  float* Cm, int M, int K, int Nc, int act) {
    dim3 g((unsigned)((M + 63) / 64), (unsigned)(Nc / 64));
    if (K == 768) {
      if (act) k_wmma_gemm<768, 1><<<g, 128, 0, stream>>>(A, Bp, bias, Cm, M, Nc);
      else     k_wmma_gemm<768, 0><<<g, 128, 0, stream>>>(A, Bp, bias, Cm, M, Nc);
    } else if (K == 256) {
      if (act) k_wmma_gemm<256, 1><<<g, 128, 0, stream>>>(A, Bp, bias, Cm, M, Nc);
      else     k_wmma_gemm<256, 0><<<g, 128, 0, stream>>>(A, Bp, bias, Cm, M, Nc);
    } else {
      if (act) k_wmma_gemm<128, 1><<<g, 128, 0, stream>>>(A, Bp, bias, Cm, M, Nc);
      else     k_wmma_gemm<128, 0><<<g, 128, 0, stream>>>(A, Bp, bias, Cm, M, Nc);
    }
  };

  // one-time (per call) bf16 conversions / B-packing
  conv(x, xb, (size_t)NN * IND);
  packB(lin1_W, lin1Wb, IND, CD);
  packB(lin2_W, lin2Wb, CD, CD);
  for (int l = 0; l < 2; ++l) {
    const int base = 4 + 13 * l;
    const float* Wq  = (const float*)d_in[base + 0];
    const float* Wk  = (const float*)d_in[base + 2];
    const float* Wv  = (const float*)d_in[base + 4];
    const float* Ws  = (const float*)d_in[base + 6];
    const float* gW  = (const float*)d_in[base + 8];
    const float* sW1 = (const float*)d_in[base + 10];
    for (int r = 0; r < RR; ++r) {
      packB(Wq + (size_t)r * CD * HDD, Wqb[l] + (size_t)r * CD * HDD, CD, HDD);
      packB(Wk + (size_t)r * CD * HDD, Wkb[l] + (size_t)r * CD * HDD, CD, HDD);
      packB(Wv + (size_t)r * CD * HDD, Wvb[l] + (size_t)r * CD * HDD, CD, HDD);
      packB(Ws + (size_t)r * CD * DHD, Wsb[l] + (size_t)r * CD * DHD, CD, DHD);
    }
    packB(gW, gWb[l], CD + DHD, CD);
    for (int hh = 0; hh < NH; ++hh)
      packB(sW1 + (size_t)hh * CD * CD, sW1b[l] + (size_t)hh * CD * CD, CD, CD);
  }

  // lin1 + LeakyReLU
  gemm(xb, lin1Wb, lin1_b, h, NN, IND, CD, 1);
  conv(h, hb, (size_t)NN * CD);

  for (int l = 0; l < 2; ++l) {
    const int base = 4 + 13 * l;
    const float* bq  = (const float*)d_in[base + 1];
    const float* bk  = (const float*)d_in[base + 3];
    const float* bv  = (const float*)d_in[base + 5];
    const float* bs  = (const float*)d_in[base + 7];
    const float* gb  = (const float*)d_in[base + 9];
    const float* sb1 = (const float*)d_in[base + 11];
    const float* sw2 = (const float*)d_in[base + 12];

    for (int r = 0; r < RR; ++r) {
      gemm(hb, Wqb[l] + (size_t)r * CD * HDD, bq + (size_t)r * HDD, qm, NN, CD, HDD, 0);
      gemm(hb, Wkb[l] + (size_t)r * CD * HDD, bk + (size_t)r * HDD, km, NN, CD, HDD, 0);
      gemm(hb, Wvb[l] + (size_t)r * CD * HDD, bv + (size_t)r * HDD, vm, NN, CD, HDD, 0);
      gemm(hb, Wsb[l] + (size_t)r * CD * DHD, bs + (size_t)r * DHD, U, NN, CD, DHD, 0);

      fill(segmax, -3.0e38f, (size_t)NN * NH);
      fill(segsum, 0.0f, (size_t)NN * NH);
      fill(agg, 0.0f, (size_t)NN * HDD);

      const int* src = edges + (size_t)(r * 2 + 0) * EE;
      const int* dst = edges + (size_t)(r * 2 + 1) * EE;
      k_edge_logits<<<cdivu((size_t)EE * NH, 256), 256, 0, stream>>>(src, dst, qm, km, logits, segmax);
      k_edge_exp<<<cdivu((size_t)EE * NH, 256), 256, 0, stream>>>(dst, logits, segmax, expo, segsum);
      k_edge_scatter<<<cdivu((size_t)EE * HDD, 256), 256, 0, stream>>>(src, dst, expo, segsum, vm, agg);
      k_head_mean<<<cdivu((size_t)NN * CD, 256), 256, 0, stream>>>(agg, U);

      k_pack_cat<<<cdivu((size_t)NN * CD, 256), 256, 0, stream>>>(U, hb, catb);
      gemm(catb, gWb[l], gb, T /* zlin */, NN, CD + DHD, CD, 0);
      k_gate<<<cdivu((size_t)NN * CD, 256), 256, 0, stream>>>(T, U, h, Hs, r);
    }

    // semantic attention
    conv(Hs, Hsb, (size_t)NN * RR * CD);
    fill(scoresP, 0.0f, NH * RR);
    for (int hh = 0; hh < NH; ++hh) {
      gemm(Hsb, sW1b[l] + (size_t)hh * CD * CD, sb1 + (size_t)hh * CD, T, NN * RR, CD, CD, 0);
      k_sem_score<<<cdivu((size_t)NN * RR, 256), 256, 0, stream>>>(T, sw2 + (size_t)hh * CD, scoresP, hh);
    }
    k_beta<<<1, 32, 0, stream>>>(scoresP, wr);
    k_sem_combine<<<cdivu((size_t)NN * CD, 256), 256, 0, stream>>>(Hs, wr, h);
    conv(h, hb, (size_t)NN * CD);
  }

  // lin2 + LeakyReLU, then classifier
  gemm(hb, lin2Wb, lin2_b, U, NN, CD, CD, 1);
  k_clf<<<cdivu(NN, 256), 256, 0, stream>>>(U, clf_W, clf_b, (float*)d_out);
}
